// LSTMEncoder_62491774156953
// MI455X (gfx1250) — compile-verified, matrix operand order checked by reading, not run on hardware
//
#include <hip/hip_runtime.h>

// ---------------------------------------------------------------------------
// 2-layer LSTM encoder (B=64, S=512, E=512, H=1024) for gfx1250 / MI455X.
//
// Phase 1 (parallel):  xp = X @ Wx^T as one huge bf16-WMMA GEMM per layer.
// Phase 2 (serial):    per timestep ONE fused kernel:
//                      gates = xp_t + h_{t-1} @ Wh^T  ->  LSTM cell update.
//
// Data movement uses the CDNA5 async global->LDS path
// (global_load_async_to_lds_b128 / ASYNCcnt) with ping-pong LDS buffers so
// the next K-chunk streams in while WMMAs consume the current one.
// Wh (8MB bf16) stays L2-resident across the 512-step chain (192MB L2).
// Workspace requirement: ~381 MB (xp buffer 256MB reused by both layers).
// ---------------------------------------------------------------------------

#define B_   64
#define S_   512
#define E_   512
#define H_   1024
#define G_   4096   // 4*H

typedef __attribute__((ext_vector_type(16))) __bf16 v16bf;
typedef __attribute__((ext_vector_type(8)))  __bf16 v8bf;
typedef __attribute__((ext_vector_type(8)))  float  v8f;

static __device__ inline v16bf combine16(v8bf lo, v8bf hi) {
  return __builtin_shufflevector(lo, hi, 0, 1, 2, 3, 4, 5, 6, 7,
                                 8, 9, 10, 11, 12, 13, 14, 15);
}

static __device__ inline unsigned short f32_to_bf16_bits(float f) {
  unsigned int u = __float_as_uint(f);
  unsigned int r = u + 0x7FFFu + ((u >> 16) & 1u);   // round-to-nearest-even
  return (unsigned short)(r >> 16);
}

static __device__ inline float bf16_bits_to_f32(unsigned short b) {
  return __uint_as_float(((unsigned int)b) << 16);
}

static __device__ inline float sigmoidf_fast(float x) {
  return 1.0f / (1.0f + __expf(-x));
}

// CDNA5 async copy: 16B global -> LDS per lane, tracked by ASYNCcnt.
static __device__ inline void async_ld16(void* lds_dst, const void* gsrc) {
  asm volatile("global_load_async_to_lds_b128 %0, %1, off"
               :: "v"((unsigned)(uintptr_t)lds_dst), "v"(gsrc)
               : "memory");
}
static __device__ inline void wait_async0() {
  asm volatile("s_wait_asynccnt 0" ::: "memory");
}

// ---------------------------------------------------------------------------
// Embedding gather: emb_bf[s][b][e] = bf16(E[x[b][s]][e]).  Rows are (s*B+b).
// ---------------------------------------------------------------------------
__global__ __launch_bounds__(128) void embed_gather_bf16(
    const int* __restrict__ x, const float* __restrict__ Etab,
    unsigned short* __restrict__ emb) {
  int bs = blockIdx.x;            // s*B + b
  int s  = bs >> 6;
  int b  = bs & 63;
  int tok = x[b * S_ + s];
  const float* row = Etab + (size_t)tok * E_;
  unsigned short* dst = emb + (size_t)bs * E_;
  for (int e = threadIdx.x; e < E_; e += 128)
    dst[e] = f32_to_bf16_bits(row[e]);
}

__global__ __launch_bounds__(256) void cvt_f32_bf16(
    const float* __restrict__ src, unsigned short* __restrict__ dst, int n) {
  int i = blockIdx.x * 256 + threadIdx.x;
  if (i < n) dst[i] = f32_to_bf16_bits(src[i]);
}

__global__ __launch_bounds__(256) void zero_u16(unsigned short* p, int n) {
  int i = blockIdx.x * 256 + threadIdx.x;
  if (i < n) p[i] = 0;
}

__global__ __launch_bounds__(256) void zero_f32(float* p, int n) {
  int i = blockIdx.x * 256 + threadIdx.x;
  if (i < n) p[i] = 0.0f;
}

// ---------------------------------------------------------------------------
// Input projection: XP[M=32768][4096] = X[M][KX] @ W[4096][KX]^T  (bf16 out).
// Grid (64, 512): block = 64 rows x 64 cols, 4 waves, async ping-pong LDS.
// Steady-state loop is branch-free (first/last chunks peeled).
// ---------------------------------------------------------------------------
template <int KX>
__global__ __launch_bounds__(128) void proj_gemm(
    const unsigned short* __restrict__ X,
    const unsigned short* __restrict__ W,
    unsigned short* __restrict__ XP) {
  __shared__ struct {
    alignas(16) unsigned short A[64][48];    // 96B rows: every frag 16B-aligned
    alignas(16) unsigned short Bp[64][48];
  } st[2];

  const int tid  = threadIdx.x;
  const int wave = tid >> 5;
  const int lane = tid & 31;
  const int half = lane >> 4;
  const int lrow = lane & 15;
  const int nbb  = blockIdx.x * 64;
  const int mbb  = blockIdx.y * 64;

  v8f acc[4] = {};

  auto stage = [&](int pp, int it) {
    const int k0 = it * 32;
#pragma unroll
    for (int c = tid; c < 256; c += 128) {          // A panel 64x32
      int row = c >> 2, sub = c & 3;
      async_ld16(&st[pp].A[row][sub * 8],
                 X + (size_t)(mbb + row) * KX + k0 + sub * 8);
    }
#pragma unroll
    for (int c = tid; c < 256; c += 128) {          // B panel 64x32
      int row = c >> 2, sub = c & 3;
      async_ld16(&st[pp].Bp[row][sub * 8],
                 W + (size_t)(nbb + row) * KX + k0 + sub * 8);
    }
  };

  auto compute = [&](int pp) {
    v8bf alo = *(const v8bf*)&st[pp].A[wave * 16 + lrow][half * 8];
    v8bf ahi = *(const v8bf*)&st[pp].A[wave * 16 + lrow][16 + half * 8];
    v16bf af = combine16(alo, ahi);
#pragma unroll
    for (int jn = 0; jn < 4; ++jn) {
      v8bf blo = *(const v8bf*)&st[pp].Bp[jn * 16 + lrow][half * 8];
      v8bf bhi = *(const v8bf*)&st[pp].Bp[jn * 16 + lrow][16 + half * 8];
      acc[jn] = __builtin_amdgcn_wmma_f32_16x16x32_bf16(
          false, af, false, combine16(blo, bhi), (short)0, acc[jn],
          false, false);
    }
  };

  constexpr int NIT = KX / 32;                      // 16 or 32 (even, >= 4)
  stage(0, 0);
  wait_async0();
  __syncthreads();
#pragma unroll 1
  for (int it = 0; it < NIT - 2; it += 2) {         // branch-free steady state
    stage(1, it + 1);
    compute(0);
    wait_async0();
    __syncthreads();
    stage(0, it + 2);
    compute(1);
    wait_async0();
    __syncthreads();
  }
  stage(1, NIT - 1);
  compute(0);
  wait_async0();
  __syncthreads();
  compute(1);

#pragma unroll
  for (int jn = 0; jn < 4; ++jn)
#pragma unroll
    for (int r = 0; r < 8; ++r) {
      int m = mbb + wave * 16 + (half ? 8 + r : r);
      XP[(size_t)m * G_ + nbb + jn * 16 + lrow] = f32_to_bf16_bits(acc[jn][r]);
    }
}

// ---------------------------------------------------------------------------
// Fused recurrent step:  gates = xp_t + Hprev @ Wh^T, then LSTM cell update.
// Grid 64 blocks (one per 16-wide h-slice), 128 threads = 4 waves.
// Wave g = gate g's 16 columns for all 64 batch rows; K unrolled x2 per
// chunk; async ping-pong LDS staging; gate tiles meet in LDS for the cell.
// ---------------------------------------------------------------------------
__global__ __launch_bounds__(128) void lstm_step_fused(
    const unsigned short* __restrict__ Hp,   // [64][1024] bf16
    const unsigned short* __restrict__ Wh,   // [4096][1024] bf16
    const unsigned short* __restrict__ xp_t, // [64][4096] bf16 slice
    const float* __restrict__ bias,          // [4096]
    float* __restrict__ cbuf,                // [64][1024] fp32 (persistent)
    unsigned short* __restrict__ h_bf,       // [64][1024] bf16 dest
    float* __restrict__ outp,                // null or (b,h)->outp+b*S*H+h
    float* __restrict__ hstate,              // null or [64*1024]
    float* __restrict__ cstate) {            // null or [64*1024]
  __shared__ union {
    struct {
      alignas(16) unsigned short A[64][72];  // 144B rows: frags 16B-aligned
      alignas(16) unsigned short Bt[4][16][72];
    } st[2];
    float g[4][64][17];                      // gate tiles for cell update
  } sh;

  const int tid  = threadIdx.x;
  const int wave = tid >> 5;                 // gate index 0..3
  const int lane = tid & 31;
  const int half = lane >> 4;
  const int lrow = lane & 15;
  const int jh   = blockIdx.x;               // h-slice 0..63
  const int nbase = wave * H_ + jh * 16;     // column base of gate `wave`

  // Warm the xp_t slice this block needs at the end (overlaps the GEMM).
#pragma unroll
  for (int mt = 0; mt < 4; ++mt)
    __builtin_prefetch(xp_t + (size_t)(mt * 16 + lrow) * G_ + nbase, 0, 3);

  v8f acc[4] = {};                           // one 16x16 tile per M-tile

  auto stage = [&](int pp, int it) {
    const int k0 = it * 64;
#pragma unroll
    for (int c = tid; c < 512; c += 128) {   // A: 64 rows x 64 K
      int row = c >> 3, sub = c & 7;
      async_ld16(&sh.st[pp].A[row][sub * 8],
                 Hp + (size_t)row * H_ + k0 + sub * 8);
    }
#pragma unroll
    for (int c = tid; c < 512; c += 128) {   // B: 4 gate strips x 16 x 64 K
      int strip = c >> 7, row = (c >> 3) & 15, sub = c & 7;
      async_ld16(&sh.st[pp].Bt[strip][row][sub * 8],
                 Wh + (size_t)(strip * H_ + jh * 16 + row) * H_ + k0 + sub * 8);
    }
  };

  auto compute = [&](int pp) {
#pragma unroll
    for (int ks = 0; ks < 2; ++ks) {
      const int kb = ks * 32;
      v8bf blo = *(const v8bf*)&sh.st[pp].Bt[wave][lrow][kb + half * 8];
      v8bf bhi = *(const v8bf*)&sh.st[pp].Bt[wave][lrow][kb + 16 + half * 8];
      v16bf bf = combine16(blo, bhi);
#pragma unroll
      for (int mt = 0; mt < 4; ++mt) {
        v8bf alo = *(const v8bf*)&sh.st[pp].A[mt * 16 + lrow][kb + half * 8];
        v8bf ahi = *(const v8bf*)&sh.st[pp].A[mt * 16 + lrow][kb + 16 + half * 8];
        acc[mt] = __builtin_amdgcn_wmma_f32_16x16x32_bf16(
            false, combine16(alo, ahi), false, bf, (short)0, acc[mt],
            false, false);
      }
    }
  };

  constexpr int NIT = H_ / 64;               // 16
  stage(0, 0);
  wait_async0();
  __syncthreads();
#pragma unroll 1
  for (int it = 0; it < NIT - 2; it += 2) {  // branch-free steady state
    stage(1, it + 1);
    compute(0);
    wait_async0();
    __syncthreads();
    stage(0, it + 2);
    compute(1);
    wait_async0();
    __syncthreads();
  }
  stage(1, NIT - 1);
  compute(0);
  wait_async0();
  __syncthreads();
  compute(1);

  // Add the precomputed input projection and stash gate tiles in LDS.
  // C/D layout: lanes 0-15 -> M=r, N=lane; lanes 16-31 -> M=8+r, N=lane-16.
  __syncthreads();                           // st[] -> g[] union reuse
#pragma unroll
  for (int mt = 0; mt < 4; ++mt)
#pragma unroll
    for (int r = 0; r < 8; ++r) {
      int m = mt * 16 + (half ? 8 + r : r);
      sh.g[wave][m][lrow] =
          acc[mt][r] + bf16_bits_to_f32(xp_t[(size_t)m * G_ + nbase + lrow]);
    }
  __syncthreads();

  // Fused LSTM cell update for this h-slice: 64 batch x 16 cols.
#pragma unroll
  for (int c = tid; c < 1024; c += 128) {
    int b = c >> 4, hc = c & 15;
    int hfull = jh * 16 + hc;
    float iv = sigmoidf_fast(sh.g[0][b][hc] + bias[hfull]);
    float fv = sigmoidf_fast(sh.g[1][b][hc] + bias[H_ + hfull]);
    float gv = tanhf        (sh.g[2][b][hc] + bias[2 * H_ + hfull]);
    float ov = sigmoidf_fast(sh.g[3][b][hc] + bias[3 * H_ + hfull]);
    size_t bh = (size_t)b * H_ + hfull;
    float cn = fv * cbuf[bh] + iv * gv;
    float hn = ov * tanhf(cn);
    cbuf[bh] = cn;
    h_bf[bh] = f32_to_bf16_bits(hn);
    if (outp)   outp[(size_t)b * ((size_t)S_ * H_) + hfull] = hn;
    if (hstate) { hstate[bh] = hn; cstate[bh] = cn; }
  }
}

// ---------------------------------------------------------------------------
// Orchestration. Workspace layout:
//   emb_bf 32MB | out0_bf 64MB | xp 256MB (reused by both layers)
//   Wx0 4MB | Wh0/Wx1/Wh1 8MB each | h1b/zeroh 128KB | cbuf 256KB  (~381MB)
// ---------------------------------------------------------------------------
extern "C" void kernel_launch(void* const* d_in, const int* in_sizes, int n_in,
                              void* d_out, int out_size, void* d_ws, size_t ws_size,
                              hipStream_t stream) {
  (void)in_sizes; (void)n_in; (void)out_size; (void)ws_size;

  const int*   x    = (const int*)  d_in[0];
  /* lengths d_in[1] unused: reference math ignores it */
  const float* Etab = (const float*)d_in[2];
  const float* Wx0  = (const float*)d_in[3];
  const float* Wh0  = (const float*)d_in[4];
  const float* b0   = (const float*)d_in[5];
  const float* Wx1  = (const float*)d_in[6];
  const float* Wh1  = (const float*)d_in[7];
  const float* b1   = (const float*)d_in[8];
  float* out = (float*)d_out;

  unsigned short* emb_bf  = (unsigned short*)d_ws;               // S*B*E
  unsigned short* out0_bf = emb_bf  + (size_t)S_ * B_ * E_;      // S*B*H
  unsigned short* xp      = out0_bf + (size_t)S_ * B_ * H_;      // S*B*G
  unsigned short* Wx0b    = xp      + (size_t)S_ * B_ * G_;      // G*E
  unsigned short* Wh0b    = Wx0b    + (size_t)G_ * E_;           // G*H
  unsigned short* Wx1b    = Wh0b    + (size_t)G_ * H_;           // G*H
  unsigned short* Wh1b    = Wx1b    + (size_t)G_ * H_;           // G*H
  unsigned short* h1b     = Wh1b    + (size_t)G_ * H_;           // B*H
  unsigned short* zeroh   = h1b     + (size_t)B_ * H_;           // B*H
  float*          cbuf    = (float*)(zeroh + (size_t)B_ * H_);   // B*H

  const size_t OUT1 = (size_t)B_ * S_ * H_;
  const size_t BH   = (size_t)B_ * H_;

  // ---- prep ----
  embed_gather_bf16<<<S_ * B_, 128, 0, stream>>>(x, Etab, emb_bf);
  {
    int n0 = G_ * E_, n1 = G_ * H_;
    cvt_f32_bf16<<<(n0 + 255) / 256, 256, 0, stream>>>(Wx0, Wx0b, n0);
    cvt_f32_bf16<<<(n1 + 255) / 256, 256, 0, stream>>>(Wh0, Wh0b, n1);
    cvt_f32_bf16<<<(n1 + 255) / 256, 256, 0, stream>>>(Wx1, Wx1b, n1);
    cvt_f32_bf16<<<(n1 + 255) / 256, 256, 0, stream>>>(Wh1, Wh1b, n1);
  }
  zero_u16<<<(int)((BH + 255) / 256), 256, 0, stream>>>(zeroh, (int)BH);

  dim3 pgrid(64, 512);   // N blocks x M blocks

  // ---- layer 0 ----
  proj_gemm<E_><<<pgrid, 128, 0, stream>>>(emb_bf, Wx0b, xp);
  zero_f32<<<(int)((BH + 255) / 256), 256, 0, stream>>>(cbuf, (int)BH);
  for (int t = 0; t < S_; ++t) {
    const unsigned short* hprev = t ? out0_bf + (size_t)(t - 1) * BH : zeroh;
    bool last = (t == S_ - 1);
    lstm_step_fused<<<64, 128, 0, stream>>>(
        hprev, Wh0b, xp + (size_t)t * B_ * G_, b0, cbuf,
        out0_bf + (size_t)t * BH,
        nullptr,
        last ? out + OUT1          : nullptr,   // h_stack[0]
        last ? out + OUT1 + 2 * BH : nullptr);  // c_stack[0]
  }

  // ---- layer 1 ----
  proj_gemm<H_><<<pgrid, 128, 0, stream>>>(out0_bf, Wx1b, xp);
  zero_f32<<<(int)((BH + 255) / 256), 256, 0, stream>>>(cbuf, (int)BH);
  for (int t = 0; t < S_; ++t) {
    const unsigned short* hprev = t ? h1b : zeroh;
    bool last = (t == S_ - 1);
    lstm_step_fused<<<64, 128, 0, stream>>>(
        hprev, Wh1b, xp + (size_t)t * B_ * G_, b1, cbuf,
        h1b,
        out + (size_t)t * H_,                   // out1[b][t][:]
        last ? out + OUT1 + BH     : nullptr,   // h_stack[1]
        last ? out + OUT1 + 3 * BH : nullptr);  // c_stack[1]
  }
}
